// SE3Transformer_49761491091512
// MI455X (gfx1250) — compile-verified, matrix operand order checked by reading, not run on hardware
//
#include <hip/hip_runtime.h>
#include <math.h>

// ---------------- types ----------------
typedef __bf16 bf16;
typedef __attribute__((ext_vector_type(16))) __bf16 v16bf;
typedef __attribute__((ext_vector_type(8)))  float  v8f;

#define NND 1024
#define C0 128
#define C1 32
#define ED 64
#define HH 4
#define LL 2
#define KK 16
#define RC 384          // 2*C0 + 4*C1
#define EE (NND*KK)     // 16384 edges

__device__ __forceinline__ float gelu_f(float x) {
    float x3 = x * x * x;
    return 0.5f * x * (1.0f + tanhf(0.7978845608028654f * (x + 0.044715f * x3)));
}

// ---------------- generic bf16 WMMA GEMM: C[M,Nc] = A[M,Kc] * W[Nc,Kc]^T ----
// A row-major [M,Kc] bf16, W row-major [Nc,Kc] bf16 (einsum 'oc,mc->mo'),
// C row-major [M,Nc] f32. Requirements: M%16==0, Nc%32==0, Kc%32==0.
// Block = 128 threads (4 waves). Wave w owns a 16x32 output slab: column tiles
// 2w and 2w+1 of this block's 128-col stripe, sharing one A fragment per
// K-step (register-level A reuse -> 2 WMMAs per A load, no branches).
__global__ void __launch_bounds__(128)
gemm_bf16_wmma(const bf16* __restrict__ A, const bf16* __restrict__ W,
               float* __restrict__ C, int M, int Nc, int Kc) {
    const int wave = threadIdx.x >> 5;
    const int lane = threadIdx.x & 31;
    const int mt  = blockIdx.x;
    const int nt0 = (blockIdx.y * 4 + wave) * 2;   // Nc%32==0 -> tiles nt0, nt0+1 both valid
    if (nt0 * 16 >= Nc) return;
    const int arow  = mt * 16 + (lane & 15);
    const int wrow0 = nt0 * 16 + (lane & 15);
    const int khalf = (lane >> 4) << 3;            // 0 or 8
    v8f acc0 = {}, acc1 = {};
    for (int k0 = 0; k0 < Kc; k0 += 32) {
        const bf16* pa  = A + (size_t)arow * Kc + k0 + khalf;
        const bf16* pw0 = W + (size_t)wrow0 * Kc + k0 + khalf;
        const bf16* pw1 = pw0 + (size_t)16 * Kc;
        v16bf a, b0, b1;
#pragma unroll
        for (int i = 0; i < 8; ++i) { a[i] = pa[i];   a[8 + i] = pa[16 + i]; }
#pragma unroll
        for (int i = 0; i < 8; ++i) { b0[i] = pw0[i]; b0[8 + i] = pw0[16 + i]; }
#pragma unroll
        for (int i = 0; i < 8; ++i) { b1[i] = pw1[i]; b1[8 + i] = pw1[16 + i]; }
        acc0 = __builtin_amdgcn_wmma_f32_16x16x32_bf16(
            false, a, false, b0, (short)0, acc0, false, false);
        acc1 = __builtin_amdgcn_wmma_f32_16x16x32_bf16(
            false, a, false, b1, (short)0, acc1, false, false);
    }
    // C/D layout: VGPR j -> (M = j + 8*(lane>>4), N = lane&15)
    float* pc0 = C + ((size_t)mt * 16 + ((lane >> 4) << 3)) * Nc + nt0 * 16 + (lane & 15);
    float* pc1 = pc0 + 16;
#pragma unroll
    for (int j = 0; j < 8; ++j) pc0[(size_t)j * Nc] = acc0[j];
#pragma unroll
    for (int j = 0; j < 8; ++j) pc1[(size_t)j * Nc] = acc1[j];
}

// ---------------- elementwise helpers ----------------
__global__ void cvt_f32_bf16_k(const float* __restrict__ in, bf16* __restrict__ out, int n) {
    int i = blockIdx.x * blockDim.x + threadIdx.x;
    if (i < n) out[i] = (bf16)in[i];
}

__global__ void gelu_cvt_k(const float* __restrict__ in, bf16* __restrict__ out, int n) {
    int i = blockIdx.x * blockDim.x + threadIdx.x;
    if (i < n) out[i] = (bf16)gelu_f(in[i]);
}

__global__ void axpy_k(float* __restrict__ y, const float* __restrict__ x,
                       const float* __restrict__ alpha, int aidx, int n) {
    int i = blockIdx.x * blockDim.x + threadIdx.x;
    if (i < n) y[i] += alpha[aidx] * x[i];
}

// f1[n,c,m] += alpha * d[(n*3+m)*C1 + c]
__global__ void axpy_f1_k(float* __restrict__ f1, const float* __restrict__ d,
                          const float* __restrict__ alpha, int aidx, int N) {
    int i = blockIdx.x * blockDim.x + threadIdx.x;
    if (i >= N * C1 * 3) return;
    int n = i / (C1 * 3), rem = i % (C1 * 3);
    int c = rem / 3, m = rem % 3;
    f1[i] += alpha[aidx] * d[((size_t)(n * 3 + m)) * C1 + c];
}

// ---------------- kNN graph (CA atoms), y1 basis, radius mask ----------------
__global__ void knn_k(const float* __restrict__ coords, int* __restrict__ idx,
                      float* __restrict__ y1, float* __restrict__ mask, int N) {
    int i = blockIdx.x * blockDim.x + threadIdx.x;
    if (i >= N) return;
    const float cx = coords[i * 12 + 3], cy = coords[i * 12 + 4], cz = coords[i * 12 + 5];
    int sel[KK];
    for (int k = 0; k < KK; ++k) {
        float bestd = 3.0e38f; int bestj = 0;
        for (int j = 0; j < N; ++j) {
            if (j == i) continue;
            bool taken = false;
            for (int t = 0; t < k; ++t) if (sel[t] == j) { taken = true; break; }
            if (taken) continue;
            float dx = coords[j * 12 + 3] - cx;
            float dy = coords[j * 12 + 4] - cy;
            float dz = coords[j * 12 + 5] - cz;
            float d = dx * dx + dy * dy + dz * dz;
            if (d < bestd) { bestd = d; bestj = j; }
        }
        sel[k] = bestj;
        idx[i * KK + k] = bestj;
        float dx = coords[bestj * 12 + 3] - cx;
        float dy = coords[bestj * 12 + 4] - cy;
        float dz = coords[bestj * 12 + 5] - cz;
        float nrm = sqrtf(dx * dx + dy * dy + dz * dz);
        mask[i * KK + k] = (sqrtf(fmaxf(bestd, 0.0f)) <= 16.0f) ? 1.0f : 0.0f;
        float inv = 1.0f / (nrm + 1e-6f);
        y1[(i * KK + k) * 3 + 0] = dx * inv;
        y1[(i * KK + k) * 3 + 1] = dy * inv;
        y1[(i * KK + k) * 3 + 2] = dz * inv;
    }
}

// ---------------- edges = LN(gather(pair_feats, idx)) ----------------
__global__ void edges_k(const float* __restrict__ pair, const int* __restrict__ idx,
                        const float* __restrict__ g, const float* __restrict__ b,
                        float* __restrict__ edges, bf16* __restrict__ edgesb, int N) {
    int e = blockIdx.x * blockDim.x + threadIdx.x;
    if (e >= N * KK) return;
    int n = e >> 4;
    int j = idx[e];
    const float* p = pair + ((size_t)n * N + j) * ED;
    float m = 0.0f;
    for (int d = 0; d < ED; ++d) m += p[d];
    m *= (1.0f / ED);
    float v = 0.0f;
    for (int d = 0; d < ED; ++d) { float t = p[d] - m; v += t * t; }
    v *= (1.0f / ED);
    float r = rsqrtf(v + 1e-5f);
    for (int d = 0; d < ED; ++d) {
        float t = (p[d] - m) * r * g[d] + b[d];
        edges[(size_t)e * ED + d] = t;
        edgesb[(size_t)e * ED + d] = (bf16)t;
    }
}

// ---------------- degree-1 input projection: f1 = Wi1 @ (stack - CA) -------
__global__ void proj_f1_k(const float* __restrict__ coords, const float* __restrict__ Wi1,
                          float* __restrict__ f1, int N) {
    int n = blockIdx.x * blockDim.x + threadIdx.x;
    if (n >= N) return;
    const float* cb = coords + n * 12;
    float fin[3][3];
    for (int m = 0; m < 3; ++m) {
        float ca = cb[3 + m];
        fin[0][m] = cb[0 + m] - ca;   // N atom
        fin[1][m] = cb[6 + m] - ca;   // C atom
        fin[2][m] = cb[9 + m] - ca;   // O atom
    }
    for (int o = 0; o < C1; ++o)
        for (int m = 0; m < 3; ++m) {
            float s = 0.0f;
            for (int c = 0; c < 3; ++c) s += Wi1[o * 3 + c] * fin[c][m];
            f1[(size_t)n * (C1 * 3) + o * 3 + m] = s;
        }
}

// ---------------- fiber norm (deg0 LN+gelu, deg1 phase-preserving) ---------
// writes x0f [N,C0], x0b bf16 [N,C0], x1f [N,C1,3], x1b bf16 rows (n*3+m) cols C1
__global__ void fibernorm_k(const float* __restrict__ f0, const float* __restrict__ f1,
                            const float* __restrict__ g0, const float* __restrict__ b0,
                            const float* __restrict__ g1, const float* __restrict__ b1,
                            float* __restrict__ x0f, bf16* __restrict__ x0b,
                            float* __restrict__ x1f, bf16* __restrict__ x1b, int N) {
    int n = blockIdx.x * blockDim.x + threadIdx.x;
    if (n >= N) return;
    const float* p0 = f0 + (size_t)n * C0;
    float m = 0.0f;
    for (int c = 0; c < C0; ++c) m += p0[c];
    m *= (1.0f / C0);
    float v = 0.0f;
    for (int c = 0; c < C0; ++c) { float d = p0[c] - m; v += d * d; }
    v *= (1.0f / C0);
    float r = rsqrtf(v + 1e-5f);
    for (int c = 0; c < C0; ++c) {
        float t = gelu_f((p0[c] - m) * r * g0[c] + b0[c]);
        x0f[(size_t)n * C0 + c] = t;
        x0b[(size_t)n * C0 + c] = (bf16)t;
    }
    const float* p1 = f1 + (size_t)n * (C1 * 3);
    float n1[C1];
    float m1 = 0.0f;
    for (int c = 0; c < C1; ++c) {
        float a0 = p1[c * 3], a1 = p1[c * 3 + 1], a2 = p1[c * 3 + 2];
        n1[c] = sqrtf(a0 * a0 + a1 * a1 + a2 * a2);
        m1 += n1[c];
    }
    m1 *= (1.0f / C1);
    float v1 = 0.0f;
    for (int c = 0; c < C1; ++c) { float d = n1[c] - m1; v1 += d * d; }
    v1 *= (1.0f / C1);
    float r1 = rsqrtf(v1 + 1e-5f);
    for (int c = 0; c < C1; ++c) {
        float t = gelu_f((n1[c] - m1) * r1 * g1[c] + b1[c]);
        float s = t / (n1[c] + 1e-6f);
        for (int mm = 0; mm < 3; ++mm) {
            float val = p1[c * 3 + mm] * s;
            x1f[(size_t)n * (C1 * 3) + c * 3 + mm] = val;
            x1b[((size_t)(n * 3 + mm)) * C1 + c] = (bf16)val;
        }
    }
}

// ---------------- TFN basis assembly (builds GEMM inputs for K and V) ------
// in0 [E, C0+C1] bf16; in1 rows (e*3+m), cols C0+3*C1 bf16
__global__ void build_in_k(const int* __restrict__ idx, const float* __restrict__ y1,
                           const float* __restrict__ x0f, const float* __restrict__ x1f,
                           const float* __restrict__ rk, const float* __restrict__ rv,
                           bf16* __restrict__ in0K, bf16* __restrict__ in1K,
                           bf16* __restrict__ in0V, bf16* __restrict__ in1V, int E) {
    int e = blockIdx.x * blockDim.x + threadIdx.x;
    if (e >= E) return;
    int j = idx[e];
    float yv[3] = { y1[e * 3], y1[e * 3 + 1], y1[e * 3 + 2] };
    const float* xj1 = x1f + (size_t)j * (C1 * 3);
    const float* xj0 = x0f + (size_t)j * C0;
    float dot1[C1];
    for (int c = 0; c < C1; ++c)
        dot1[c] = xj1[c * 3] * yv[0] + xj1[c * 3 + 1] * yv[1] + xj1[c * 3 + 2] * yv[2];
    for (int path = 0; path < 2; ++path) {
        const float* r = (path ? rv : rk) + (size_t)e * RC;
        bf16* i0 = (path ? in0V : in0K) + (size_t)e * (C0 + C1);
        bf16* i1 = (path ? in1V : in1K) + (size_t)e * 3 * (C0 + 3 * C1);
        const int S = C0 + 3 * C1;  // 224 row stride in in1
        for (int c = 0; c < C0; ++c) i0[c] = (bf16)(r[c] * xj0[c]);
        for (int c = 0; c < C1; ++c) i0[C0 + c] = (bf16)(r[C0 + c] * dot1[c]);
        for (int c = 0; c < C0; ++c) {                       // path 0->1
            float t = r[160 + c] * xj0[c];
            for (int m = 0; m < 3; ++m) i1[m * S + c] = (bf16)(t * yv[m]);
        }
        for (int c = 0; c < C1; ++c)                          // path 1->1 (l=0)
            for (int m = 0; m < 3; ++m)
                i1[m * S + C0 + c] = (bf16)(r[288 + c] * xj1[c * 3 + m]);
        for (int c = 0; c < C1; ++c) {                        // path 1->1 cross
            float a0 = xj1[c * 3], a1 = xj1[c * 3 + 1], a2 = xj1[c * 3 + 2];
            float cr[3] = { a1 * yv[2] - a2 * yv[1],
                            a2 * yv[0] - a0 * yv[2],
                            a0 * yv[1] - a1 * yv[0] };
            for (int m = 0; m < 3; ++m)
                i1[m * S + C0 + C1 + c] = (bf16)(r[320 + c] * cr[m]);
        }
        for (int c = 0; c < C1; ++c)                          // path 1->1 (l=2)
            for (int m = 0; m < 3; ++m)
                i1[m * S + C0 + 2 * C1 + c] =
                    (bf16)(r[352 + c] * (yv[m] * dot1[c] - xj1[c * 3 + m] * (1.0f / 3.0f)));
    }
}

// ---------------- attention over K=16 neighbors, per (node, head) ----------
__global__ void attn_k(const float* __restrict__ q0, const float* __restrict__ q1,
                       const float* __restrict__ k0, const float* __restrict__ k1,
                       const float* __restrict__ v0, const float* __restrict__ v1,
                       const float* __restrict__ edges, const float* __restrict__ Wb,
                       const float* __restrict__ mask,
                       bf16* __restrict__ o0b, bf16* __restrict__ o1b, int N) {
    int t = blockIdx.x * blockDim.x + threadIdx.x;
    if (t >= N * HH) return;
    int n = t / HH, h = t % HH;
    const int dh0 = C0 / HH, dh1 = C1 / HH;
    const float scale = 0.13363062095621219f;   // 1/sqrt(32 + 3*8)
    float logits[KK];
    float lmax = -3.0e38f;
    for (int k = 0; k < KK; ++k) {
        int e = n * KK + k;
        float s = 0.0f;
        for (int c = 0; c < dh0; ++c)
            s += q0[(size_t)n * C0 + h * dh0 + c] * k0[(size_t)e * C0 + h * dh0 + c];
        for (int m = 0; m < 3; ++m)
            for (int c = 0; c < dh1; ++c)
                s += q1[((size_t)(n * 3 + m)) * C1 + h * dh1 + c] *
                     k1[((size_t)(e * 3 + m)) * C1 + h * dh1 + c];
        s *= scale;
        float bb = 0.0f;
        for (int d = 0; d < ED; ++d) bb += Wb[h * ED + d] * edges[(size_t)e * ED + d];
        s += bb;
        if (mask[e] == 0.0f) s = -1e9f;
        logits[k] = s;
        lmax = fmaxf(lmax, s);
    }
    float den = 0.0f;
    for (int k = 0; k < KK; ++k) { logits[k] = expf(logits[k] - lmax); den += logits[k]; }
    float inv = 1.0f / den;
    for (int c = 0; c < dh0; ++c) {
        float s = 0.0f;
        for (int k = 0; k < KK; ++k)
            s += logits[k] * v0[((size_t)(n * KK + k)) * C0 + h * dh0 + c];
        o0b[(size_t)n * C0 + h * dh0 + c] = (bf16)(s * inv);
    }
    for (int m = 0; m < 3; ++m)
        for (int c = 0; c < dh1; ++c) {
            float s = 0.0f;
            for (int k = 0; k < KK; ++k)
                s += logits[k] * v1[((size_t)((n * KK + k) * 3 + m)) * C1 + h * dh1 + c];
            o1b[((size_t)(n * 3 + m)) * C1 + h * dh1 + c] = (bf16)(s * inv);
        }
}

// ---------------- FF deg-1 magnitude gate ----------------
__global__ void ff_gate_k(const float* __restrict__ h1a, bf16* __restrict__ h1ab, int N) {
    int t = blockIdx.x * blockDim.x + threadIdx.x;
    const int HC = 4 * C1;   // 128
    if (t >= N * HC) return;
    int n = t / HC, h = t % HC;
    float a0 = h1a[((size_t)(n * 3 + 0)) * HC + h];
    float a1 = h1a[((size_t)(n * 3 + 1)) * HC + h];
    float a2 = h1a[((size_t)(n * 3 + 2)) * HC + h];
    float nh = sqrtf(a0 * a0 + a1 * a1 + a2 * a2);
    float s = gelu_f(nh) / (nh + 1e-6f);
    h1ab[((size_t)(n * 3 + 0)) * HC + h] = (bf16)(a0 * s);
    h1ab[((size_t)(n * 3 + 1)) * HC + h] = (bf16)(a1 * s);
    h1ab[((size_t)(n * 3 + 2)) * HC + h] = (bf16)(a2 * s);
}

// ---------------- output coords head ----------------
__global__ void coords_out_k(const float* __restrict__ coords, const float* __restrict__ Wp1,
                             const float* __restrict__ f1, float* __restrict__ out, int N) {
    int n = blockIdx.x * blockDim.x + threadIdx.x;
    if (n >= N) return;
    for (int o = 0; o < 3; ++o)
        for (int m = 0; m < 3; ++m) {
            float s = coords[n * 12 + 3 + m];   // CA
            for (int c = 0; c < C1; ++c)
                s += Wp1[o * C1 + c] * f1[(size_t)n * (C1 * 3) + c * 3 + m];
            out[(size_t)n * 9 + o * 3 + m] = s;
        }
}

// ---------------- host ----------------
extern "C" void kernel_launch(void* const* d_in, const int* in_sizes, int n_in,
                              void* d_out, int out_size, void* d_ws, size_t ws_size,
                              hipStream_t stream) {
    (void)in_sizes; (void)n_in; (void)out_size; (void)ws_size;
    const float* node_feats = (const float*)d_in[0];
    const float* pair_feats = (const float*)d_in[1];
    const float* coords     = (const float*)d_in[2];
    const float* eln_g = (const float*)d_in[3];
    const float* eln_b = (const float*)d_in[4];
    const float* Wi0 = (const float*)d_in[5];
    const float* Wi1 = (const float*)d_in[6];
    const float* ln0_g = (const float*)d_in[7];
    const float* ln0_b = (const float*)d_in[8];
    const float* n1_g = (const float*)d_in[9];
    const float* n1_b = (const float*)d_in[10];
    const float* Wq0 = (const float*)d_in[11];
    const float* Wq1 = (const float*)d_in[12];
    const float* WrK = (const float*)d_in[13];
    const float* WrV = (const float*)d_in[14];
    const float* WkA = (const float*)d_in[15];
    const float* WkB = (const float*)d_in[16];
    const float* WvA = (const float*)d_in[17];
    const float* WvB = (const float*)d_in[18];
    const float* Wb  = (const float*)d_in[19];
    const float* Wo0 = (const float*)d_in[20];
    const float* Wo1 = (const float*)d_in[21];
    const float* alpha_attn = (const float*)d_in[22];
    const float* lnf0_g = (const float*)d_in[23];
    const float* lnf0_b = (const float*)d_in[24];
    const float* nf1_g = (const float*)d_in[25];
    const float* nf1_b = (const float*)d_in[26];
    const float* Wf0a = (const float*)d_in[27];
    const float* Wf0b = (const float*)d_in[28];
    const float* Wf1a = (const float*)d_in[29];
    const float* Wf1b = (const float*)d_in[30];
    const float* alpha_ff = (const float*)d_in[31];
    const float* Wp0 = (const float*)d_in[32];
    const float* Wp1 = (const float*)d_in[33];

    const int N = NND, E = EE;

    // ---- workspace bump allocator ----
    char* wp = (char*)d_ws;
    auto alloc = [&](size_t bytes) -> void* {
        void* r = (void*)wp;
        wp += (bytes + 255) & ~(size_t)255;
        return r;
    };
    // bf16 weight arena (full [L,...] arrays, indexed per layer)
    bf16* Wi0b  = (bf16*)alloc((size_t)C0 * C0 * 2);
    bf16* Wq0b  = (bf16*)alloc((size_t)LL * C0 * C0 * 2);
    bf16* Wq1b  = (bf16*)alloc((size_t)LL * C1 * C1 * 2);
    bf16* WrKb  = (bf16*)alloc((size_t)LL * RC * ED * 2);
    bf16* WrVb  = (bf16*)alloc((size_t)LL * RC * ED * 2);
    bf16* WkAb  = (bf16*)alloc((size_t)LL * C0 * (C0 + C1) * 2);
    bf16* WkBb  = (bf16*)alloc((size_t)LL * C1 * (C0 + 3 * C1) * 2);
    bf16* WvAb  = (bf16*)alloc((size_t)LL * C0 * (C0 + C1) * 2);
    bf16* WvBb  = (bf16*)alloc((size_t)LL * C1 * (C0 + 3 * C1) * 2);
    bf16* Wo0b  = (bf16*)alloc((size_t)LL * C0 * C0 * 2);
    bf16* Wo1b  = (bf16*)alloc((size_t)LL * C1 * C1 * 2);
    bf16* Wf0ab = (bf16*)alloc((size_t)LL * 4 * C0 * C0 * 2);
    bf16* Wf0bb = (bf16*)alloc((size_t)LL * C0 * 4 * C0 * 2);
    bf16* Wf1ab = (bf16*)alloc((size_t)LL * 4 * C1 * C1 * 2);
    bf16* Wf1bb = (bf16*)alloc((size_t)LL * C1 * 4 * C1 * 2);
    bf16* Wp0b  = (bf16*)alloc((size_t)C0 * C0 * 2);
    // activations / graph
    int*   idx   = (int*)alloc((size_t)E * 4);
    float* y1    = (float*)alloc((size_t)E * 3 * 4);
    float* maskb = (float*)alloc((size_t)E * 4);
    float* edges = (float*)alloc((size_t)E * ED * 4);
    bf16*  edgesb= (bf16*)alloc((size_t)E * ED * 2);
    bf16*  nodeb = (bf16*)alloc((size_t)N * C0 * 2);
    float* f0    = (float*)alloc((size_t)N * C0 * 4);
    float* f1    = (float*)alloc((size_t)N * C1 * 3 * 4);
    float* x0f   = (float*)alloc((size_t)N * C0 * 4);
    bf16*  x0b   = (bf16*)alloc((size_t)N * C0 * 2);
    float* x1f   = (float*)alloc((size_t)N * C1 * 3 * 4);
    bf16*  x1b   = (bf16*)alloc((size_t)N * 3 * C1 * 2);
    float* q0    = (float*)alloc((size_t)N * C0 * 4);
    float* q1    = (float*)alloc((size_t)N * 3 * C1 * 4);
    float* rk    = (float*)alloc((size_t)E * RC * 4);
    float* rv    = (float*)alloc((size_t)E * RC * 4);
    bf16*  in0K  = (bf16*)alloc((size_t)E * (C0 + C1) * 2);
    bf16*  in1K  = (bf16*)alloc((size_t)E * 3 * (C0 + 3 * C1) * 2);
    bf16*  in0V  = (bf16*)alloc((size_t)E * (C0 + C1) * 2);
    bf16*  in1V  = (bf16*)alloc((size_t)E * 3 * (C0 + 3 * C1) * 2);
    float* k0    = (float*)alloc((size_t)E * C0 * 4);
    float* k1    = (float*)alloc((size_t)E * 3 * C1 * 4);
    float* v0    = (float*)alloc((size_t)E * C0 * 4);
    float* v1    = (float*)alloc((size_t)E * 3 * C1 * 4);
    bf16*  o0b   = (bf16*)alloc((size_t)N * C0 * 2);
    bf16*  o1b   = (bf16*)alloc((size_t)N * 3 * C1 * 2);
    float* do0   = (float*)alloc((size_t)N * C0 * 4);
    float* do1   = (float*)alloc((size_t)N * 3 * C1 * 4);
    float* hmid  = (float*)alloc((size_t)N * 4 * C0 * 4);
    bf16*  hmidb = (bf16*)alloc((size_t)N * 4 * C0 * 2);
    float* h0    = (float*)alloc((size_t)N * C0 * 4);
    float* h1a   = (float*)alloc((size_t)N * 3 * 4 * C1 * 4);
    bf16*  h1ab  = (bf16*)alloc((size_t)N * 3 * 4 * C1 * 2);
    float* h1    = (float*)alloc((size_t)N * 3 * C1 * 4);
    bf16*  f0b   = (bf16*)alloc((size_t)N * C0 * 2);

    auto cvt = [&](const float* s, bf16* d, int n) {
        cvt_f32_bf16_k<<<(n + 255) / 256, 256, 0, stream>>>(s, d, n);
    };
    auto gemm = [&](const bf16* A, const bf16* W, float* C, int M, int Nc, int Kc) {
        // each block covers 16 rows x 128 cols (4 waves x 2 tiles)
        dim3 g(M / 16, (Nc + 127) / 128);
        gemm_bf16_wmma<<<g, 128, 0, stream>>>(A, W, C, M, Nc, Kc);
    };

    // ---- weight conversions (deterministic, every launch) ----
    cvt(Wi0, Wi0b, C0 * C0);
    cvt(Wq0, Wq0b, LL * C0 * C0);
    cvt(Wq1, Wq1b, LL * C1 * C1);
    cvt(WrK, WrKb, LL * RC * ED);
    cvt(WrV, WrVb, LL * RC * ED);
    cvt(WkA, WkAb, LL * C0 * (C0 + C1));
    cvt(WkB, WkBb, LL * C1 * (C0 + 3 * C1));
    cvt(WvA, WvAb, LL * C0 * (C0 + C1));
    cvt(WvB, WvBb, LL * C1 * (C0 + 3 * C1));
    cvt(Wo0, Wo0b, LL * C0 * C0);
    cvt(Wo1, Wo1b, LL * C1 * C1);
    cvt(Wf0a, Wf0ab, LL * 4 * C0 * C0);
    cvt(Wf0b, Wf0bb, LL * C0 * 4 * C0);
    cvt(Wf1a, Wf1ab, LL * 4 * C1 * C1);
    cvt(Wf1b, Wf1bb, LL * C1 * 4 * C1);
    cvt(Wp0, Wp0b, C0 * C0);
    cvt(node_feats, nodeb, N * C0);

    // ---- graph + edges + input projections ----
    knn_k<<<(N + 127) / 128, 128, 0, stream>>>(coords, idx, y1, maskb, N);
    edges_k<<<(E + 127) / 128, 128, 0, stream>>>(pair_feats, idx, eln_g, eln_b, edges, edgesb, N);
    proj_f1_k<<<(N + 127) / 128, 128, 0, stream>>>(coords, Wi1, f1, N);
    gemm(nodeb, Wi0b, f0, N, C0, C0);                       // f0 = Wi0 @ node_feats

    for (int l = 0; l < LL; ++l) {
        // ---- attention block ----
        fibernorm_k<<<(N + 127) / 128, 128, 0, stream>>>(
            f0, f1, ln0_g + l * C0, ln0_b + l * C0, n1_g + l * C1, n1_b + l * C1,
            x0f, x0b, x1f, x1b, N);
        gemm(x0b, Wq0b + l * C0 * C0, q0, N, C0, C0);
        gemm(x1b, Wq1b + l * C1 * C1, q1, N * 3, C1, C1);
        gemm(edgesb, WrKb + l * RC * ED, rk, E, RC, ED);
        gemm(edgesb, WrVb + l * RC * ED, rv, E, RC, ED);
        build_in_k<<<(E + 127) / 128, 128, 0, stream>>>(
            idx, y1, x0f, x1f, rk, rv, in0K, in1K, in0V, in1V, E);
        gemm(in0K, WkAb + l * C0 * (C0 + C1), k0, E, C0, C0 + C1);
        gemm(in1K, WkBb + l * C1 * (C0 + 3 * C1), k1, E * 3, C1, C0 + 3 * C1);
        gemm(in0V, WvAb + l * C0 * (C0 + C1), v0, E, C0, C0 + C1);
        gemm(in1V, WvBb + l * C1 * (C0 + 3 * C1), v1, E * 3, C1, C0 + 3 * C1);
        attn_k<<<(N * HH + 127) / 128, 128, 0, stream>>>(
            q0, q1, k0, k1, v0, v1, edges, Wb + l * HH * ED, maskb, o0b, o1b, N);
        gemm(o0b, Wo0b + l * C0 * C0, do0, N, C0, C0);
        gemm(o1b, Wo1b + l * C1 * C1, do1, N * 3, C1, C1);
        axpy_k<<<(N * C0 + 255) / 256, 256, 0, stream>>>(f0, do0, alpha_attn, l, N * C0);
        axpy_f1_k<<<(N * C1 * 3 + 255) / 256, 256, 0, stream>>>(f1, do1, alpha_attn, l, N);
        // ---- fiber feedforward ----
        fibernorm_k<<<(N + 127) / 128, 128, 0, stream>>>(
            f0, f1, lnf0_g + l * C0, lnf0_b + l * C0, nf1_g + l * C1, nf1_b + l * C1,
            x0f, x0b, x1f, x1b, N);
        gemm(x0b, Wf0ab + l * 4 * C0 * C0, hmid, N, 4 * C0, C0);
        gelu_cvt_k<<<(N * 4 * C0 + 255) / 256, 256, 0, stream>>>(hmid, hmidb, N * 4 * C0);
        gemm(hmidb, Wf0bb + l * C0 * 4 * C0, h0, N, C0, 4 * C0);
        gemm(x1b, Wf1ab + l * 4 * C1 * C1, h1a, N * 3, 4 * C1, C1);
        ff_gate_k<<<(N * 4 * C1 + 255) / 256, 256, 0, stream>>>(h1a, h1ab, N);
        gemm(h1ab, Wf1bb + l * C1 * 4 * C1, h1, N * 3, C1, 4 * C1);
        axpy_k<<<(N * C0 + 255) / 256, 256, 0, stream>>>(f0, h0, alpha_ff, l, N * C0);
        axpy_f1_k<<<(N * C1 * 3 + 255) / 256, 256, 0, stream>>>(f1, h1, alpha_ff, l, N);
    }

    // ---- output heads ----
    cvt(f0, f0b, N * C0);
    gemm(f0b, Wp0b, (float*)d_out, N, C0, C0);              // node_out [N,128]
    coords_out_k<<<(N + 127) / 128, 128, 0, stream>>>(
        coords, Wp1, f1, (float*)d_out + (size_t)N * C0, N);
}
